// fMetaNMMF_19232863552120
// MI455X (gfx1250) — matrix operand order repaired and checked by reading, not compiled, not run on hardware
//
#include <hip/hip_runtime.h>
#include <hip/hip_bf16.h>

// ---------------- problem constants ----------------
#define BSZ   4096
#define TSTEP 64
#define INSZ  144
#define HID   128
#define KPAD  160          // 144 padded to multiple of 32 for WMMA K-steps

#define CTX_TILES ((BSZ*TSTEP)/16)   // 16384
#define CUR_TILES (BSZ/16)           // 256
#define NEG_TILES ((BSZ*2)/16)       // 512
#define ENC_BLOCKS (CTX_TILES + CUR_TILES + NEG_TILES)

// ---------------- d_in indices (pytree: top-level dict order, nested dicts key-sorted) ---
#define IN_CTX    0
#define IN_CUR    1
#define IN_NEG    2
#define IN_ATTN_WQ 3
#define IN_ATTN_WV 4
#define IN_ATTN_BQ 5
#define IN_ATTN_BV 6
#define IN_CTXP_W1 7
#define IN_CTXP_W2 8
#define IN_CTXP_B1 9
#define IN_CTXP_B2 10
#define IN_CURP_W1 11
#define IN_CURP_W2 12
#define IN_CURP_B1 13
#define IN_CURP_B2 14
#define IN_DEC_W1 15
#define IN_DEC_W2 16
#define IN_DEC_W3 17
#define IN_DEC_A1 18
#define IN_DEC_A2 19
#define IN_DEC_B1 20
#define IN_DEC_B2 21
#define IN_DEC_B3 22
#define IN_DEC_BE1 23
#define IN_DEC_BE2 24
#define IN_DEC_G1 25
#define IN_DEC_G2 26
#define IN_ENC_W1 27
#define IN_ENC_W2 28
#define IN_ENC_W3 29
#define IN_ENC_A1 30
#define IN_ENC_A2 31
#define IN_ENC_B1 32
#define IN_ENC_B2 33
#define IN_ENC_B3 34
#define IN_ENC_BE1 35
#define IN_ENC_BE2 36
#define IN_ENC_G1 37
#define IN_ENC_G2 38
#define IN_NORM_B 43
#define IN_NORM_G 44

// ---------------- workspace layout (bytes) ----------------
#define WS_CTXSUM   0u                               // BSZ*128 f32 = 2 MB
#define WS_CURRF    2097152u                         // BSZ*128 f32
#define WS_NEGF     4194304u                         // 2*BSZ*128 f32 = 4 MB
#define WS_LOSS     8388608u                         // 1 f32 (+pad)
#define WS_EW1H     8388864u                         // 128*160 f16
#define WS_EW2H     (WS_EW1H + 40960u)               // 128*128 f16
#define WS_EW3H     (WS_EW2H + 32768u)
#define WS_WVH      (WS_EW3H + 32768u)
#define WS_DW1H     (WS_WVH  + 32768u)
#define WS_DW2H     (WS_DW1H + 32768u)
#define WS_DW3H     (WS_DW2H + 32768u)               // 144*128 f16

typedef __attribute__((ext_vector_type(16))) _Float16 v16h;
typedef __attribute__((ext_vector_type(8)))  _Float16 v8h;
typedef __attribute__((ext_vector_type(8)))  float    v8f;

// ---------------- WMMA helpers (CDNA5 16x16x32 f16, f32 accumulate) ----------------
__device__ __forceinline__ v8f wmma16(v16h a, v16h b, v8f c) {
  return __builtin_amdgcn_wmma_f32_16x16x32_f16(false, a, false, b, (short)0, c,
                                                false, false);
}

// A fragment 16x32 (MxK) f16 from LDS, row stride ldk halfs.
// Lane l holds row m=l&15; lanes 0-15: K [k0,k0+16);
// lanes 16-31: [k0+16,k0+32) (contiguous, matching frag_b).
__device__ __forceinline__ v16h frag_a(const _Float16* A, int lane, int k0, int ldk) {
  int m = lane & 15;
  int kq = k0 + ((lane >> 4) << 4);
  v8h lo = *(const v8h*)(A + m * ldk + kq);
  v8h hi = *(const v8h*)(A + m * ldk + kq + 8);
  v16h r;
  #pragma unroll
  for (int i = 0; i < 8; ++i) { r[i] = lo[i]; r[i + 8] = hi[i]; }
  return r;
}

// B fragment 32x16 (KxN) f16 from weight stored as W[n][k] (row-major, stride ldk).
// Lane l -> column n = n0 + (l&15); 16 contiguous K halfs starting at
// k0 + (l>>4)*16.  (contiguous k order, same as frag_a; not the ISA table order)
__device__ __forceinline__ v16h frag_b(const _Float16* W, int lane, int n0, int k0, int ldk) {
  int n = n0 + (lane & 15);
  int kb = k0 + ((lane >> 4) << 4);
  return *(const v16h*)(W + (size_t)n * ldk + kb);
}

// LayerNorm + PReLU over 16 rows x 128 cols, 256 threads; writes f16 (stride 160).
// alpha = 1.0f gives plain LayerNorm (identity activation).
__device__ __forceinline__ void ln_act_256(const float* Xs, _Float16* As,
                                           const float* g, const float* be,
                                           float alpha, float* red) {
  const int tid = threadIdx.x;
  const int r = tid >> 4, c = tid & 15;
  float s = 0.f, s2 = 0.f;
  #pragma unroll
  for (int j = c; j < 128; j += 16) { float x = Xs[r * 128 + j]; s += x; s2 += x * x; }
  red[r * 16 + c] = s;
  red[256 + r * 16 + c] = s2;
  __syncthreads();
  float m = 0.f, q = 0.f;
  #pragma unroll
  for (int j = 0; j < 16; ++j) { m += red[r * 16 + j]; q += red[256 + r * 16 + j]; }
  m *= (1.f / 128.f);
  q = q * (1.f / 128.f) - m * m;
  float rs = rsqrtf(q + 1e-5f);
  #pragma unroll
  for (int j = c; j < 128; j += 16) {
    float x = (Xs[r * 128 + j] - m) * rs * g[j] + be[j];
    x = (x >= 0.f) ? x : alpha * x;
    As[r * KPAD + j] = (_Float16)x;
  }
  __syncthreads();
}

// ---------------- kernel 0: zero accumulators + convert weights to f16 ----------------
__global__ __launch_bounds__(256) void prep_kernel(
    const float* eW1, const float* eW2, const float* eW3, const float* aWv,
    const float* dW1, const float* dW2, const float* dW3,
    _Float16* eW1h, _Float16* eW2h, _Float16* eW3h, _Float16* aWvh,
    _Float16* dW1h, _Float16* dW2h, _Float16* dW3h,
    float* ctxsum, float* loss) {
  long i = (long)blockIdx.x * 256 + threadIdx.x;
  if (i == 0) loss[0] = 0.f;
  if (i < BSZ * 128) { ctxsum[i] = 0.f; return; }
  i -= BSZ * 128;
  if (i < 128 * KPAD) {                       // enc W1: pad K 144 -> 160
    int n = (int)(i / KPAD), k = (int)(i % KPAD);
    eW1h[i] = (_Float16)(k < INSZ ? eW1[n * INSZ + k] : 0.f);
    return;
  }
  i -= 128 * KPAD;
  if (i < 16384) { eW2h[i] = (_Float16)eW2[i]; return; }
  i -= 16384;
  if (i < 16384) { eW3h[i] = (_Float16)eW3[i]; return; }
  i -= 16384;
  if (i < 16384) { aWvh[i] = (_Float16)aWv[i]; return; }
  i -= 16384;
  if (i < 16384) { dW1h[i] = (_Float16)dW1[i]; return; }
  i -= 16384;
  if (i < 16384) { dW2h[i] = (_Float16)dW2[i]; return; }
  i -= 16384;
  if (i < INSZ * 128) { dW3h[i] = (_Float16)dW3[i]; return; }
}

// ---------------- kernel 1: fused encoder (+ Wv/tanh/mean for ctx rows) -------------
__global__ __launch_bounds__(256) void enc_kernel(
    const float* __restrict__ ctx, const float* __restrict__ cur,
    const float* __restrict__ neg,
    const _Float16* __restrict__ W1h, const _Float16* __restrict__ W2h,
    const _Float16* __restrict__ W3h, const _Float16* __restrict__ Wvh,
    const float* __restrict__ eb1, const float* __restrict__ eg1,
    const float* __restrict__ ebe1, const float* __restrict__ ea1,
    const float* __restrict__ eb2, const float* __restrict__ eg2,
    const float* __restrict__ ebe2, const float* __restrict__ ea2,
    const float* __restrict__ eb3, const float* __restrict__ bv,
    float* __restrict__ ctxsum, float* __restrict__ currf,
    float* __restrict__ negf) {
  __shared__ __align__(32) _Float16 As[16 * KPAD];
  __shared__ float Xs[16 * 128];
  __shared__ float red[512];
  const int tid = threadIdx.x, lane = tid & 31, wave = tid >> 5;
  const int n0 = wave * 16;
  const int tile = blockIdx.x;
  int type, t2 = 0;
  if (tile < CTX_TILES) { type = 0; t2 = tile; }
  else if (tile < CTX_TILES + CUR_TILES) { type = 1; t2 = tile - CTX_TILES; }
  else { type = 2; t2 = tile - CTX_TILES - CUR_TILES; }

  // ---- load 16x144 input rows -> f16 LDS (pad to 160) ----
  for (int i = tid; i < 16 * INSZ; i += 256) {
    int r = i / INSZ, c = i - r * INSZ;
    float x;
    if (type == 0)      x = ctx[(size_t)(t2 * 16 + r) * INSZ + c];
    else if (type == 1) x = cur[(size_t)(t2 * 16 + r) * INSZ + c];
    else { int gr = t2 * 16 + r; int b = gr >> 1, w = gr & 1;
           x = neg[(size_t)(b * 8 + w) * INSZ + c]; }
    As[r * KPAD + c] = (_Float16)x;
  }
  { int r = tid >> 4, c = tid & 15; As[r * KPAD + INSZ + c] = (_Float16)0.f; }
  __syncthreads();

  // ---- layer 1 (K = 160) ----
  v8f acc1 = {};
  #pragma unroll
  for (int kt = 0; kt < 5; ++kt)
    acc1 = wmma16(frag_a(As, lane, kt * 32, KPAD),
                  frag_b(W1h, lane, n0, kt * 32, KPAD), acc1);
  #pragma unroll
  for (int v = 0; v < 8; ++v) {
    int m = v + ((lane >> 4) << 3), n = n0 + (lane & 15);
    Xs[m * 128 + n] = acc1[v] + eb1[n];
  }
  __syncthreads();
  ln_act_256(Xs, As, eg1, ebe1, ea1[0], red);

  // ---- layer 2 (K = 128) ----
  v8f acc2 = {};
  #pragma unroll
  for (int kt = 0; kt < 4; ++kt)
    acc2 = wmma16(frag_a(As, lane, kt * 32, KPAD),
                  frag_b(W2h, lane, n0, kt * 32, 128), acc2);
  #pragma unroll
  for (int v = 0; v < 8; ++v) {
    int m = v + ((lane >> 4) << 3), n = n0 + (lane & 15);
    Xs[m * 128 + n] = acc2[v] + eb2[n];
  }
  __syncthreads();
  ln_act_256(Xs, As, eg2, ebe2, ea2[0], red);

  // ---- layer 3 (K = 128) -> encoder features in Xs ----
  v8f acc3 = {};
  #pragma unroll
  for (int kt = 0; kt < 4; ++kt)
    acc3 = wmma16(frag_a(As, lane, kt * 32, KPAD),
                  frag_b(W3h, lane, n0, kt * 32, 128), acc3);
  #pragma unroll
  for (int v = 0; v < 8; ++v) {
    int m = v + ((lane >> 4) << 3), n = n0 + (lane & 15);
    Xs[m * 128 + n] = acc3[v] + eb3[n];
  }
  __syncthreads();

  if (type == 1) {          // current features -> ws
    for (int i = tid; i < 2048; i += 256) {
      int r = i >> 7, c = i & 127;
      currf[(size_t)(t2 * 16 + r) * 128 + c] = Xs[i];
    }
    return;
  }
  if (type == 2) {          // negative features -> ws (row = b*2 + w)
    for (int i = tid; i < 2048; i += 256) {
      int r = i >> 7, c = i & 127;
      negf[(size_t)(t2 * 16 + r) * 128 + c] = Xs[i];
    }
    return;
  }

  // ---- ctx: k = tanh(Wv f + bv); accumulate sum over T into ctxsum[b] ----
  for (int i = tid; i < 2048; i += 256) {
    int r = i >> 7, c = i & 127;
    As[r * KPAD + c] = (_Float16)Xs[i];
  }
  __syncthreads();
  v8f accv = {};
  #pragma unroll
  for (int kt = 0; kt < 4; ++kt)
    accv = wmma16(frag_a(As, lane, kt * 32, KPAD),
                  frag_b(Wvh, lane, n0, kt * 32, 128), accv);
  #pragma unroll
  for (int v = 0; v < 8; ++v) {
    int m = v + ((lane >> 4) << 3), n = n0 + (lane & 15);
    Xs[m * 128 + n] = tanhf(accv[v] + bv[n]);
  }
  __syncthreads();
  if (tid < 128) {
    float s = 0.f;
    #pragma unroll
    for (int r = 0; r < 16; ++r) s += Xs[r * 128 + tid];
    int b = t2 >> 2;                         // 4 tiles of 16 per batch row (T=64)
    atomicAdd(&ctxsum[(size_t)b * 128 + tid], s);
  }
}

// ---------------- kernel 2: dec_input LN + decoder MLP3 + sigmoid ----------------
__global__ __launch_bounds__(256) void dec_kernel(
    const float* __restrict__ ctxsum, const float* __restrict__ currf,
    const _Float16* __restrict__ dW1h, const _Float16* __restrict__ dW2h,
    const _Float16* __restrict__ dW3h,
    const float* __restrict__ ng, const float* __restrict__ nb,
    const float* __restrict__ db1, const float* __restrict__ dg1,
    const float* __restrict__ dbe1, const float* __restrict__ da1,
    const float* __restrict__ db2, const float* __restrict__ dg2,
    const float* __restrict__ dbe2, const float* __restrict__ da2,
    const float* __restrict__ db3, float* __restrict__ out) {
  __shared__ __align__(32) _Float16 As[16 * KPAD];
  __shared__ float Xs[16 * 128];
  __shared__ float red[512];
  const int tid = threadIdx.x, lane = tid & 31, wave = tid >> 5;
  const int n0 = wave * 16;
  const int rb0 = blockIdx.x * 16;

  for (int i = tid; i < 2048; i += 256) {
    int r = i >> 7, c = i & 127;
    size_t g = (size_t)(rb0 + r) * 128 + c;
    Xs[i] = ctxsum[g] * (1.f / 64.f) + currf[g];   // ctx_vec + curr_feature
  }
  __syncthreads();
  ln_act_256(Xs, As, ng, nb, 1.0f, red);           // alpha=1 -> plain LN

  v8f a1 = {};
  #pragma unroll
  for (int kt = 0; kt < 4; ++kt)
    a1 = wmma16(frag_a(As, lane, kt * 32, KPAD),
                frag_b(dW1h, lane, n0, kt * 32, 128), a1);
  #pragma unroll
  for (int v = 0; v < 8; ++v) {
    int m = v + ((lane >> 4) << 3), n = n0 + (lane & 15);
    Xs[m * 128 + n] = a1[v] + db1[n];
  }
  __syncthreads();
  ln_act_256(Xs, As, dg1, dbe1, da1[0], red);

  v8f a2 = {};
  #pragma unroll
  for (int kt = 0; kt < 4; ++kt)
    a2 = wmma16(frag_a(As, lane, kt * 32, KPAD),
                frag_b(dW2h, lane, n0, kt * 32, 128), a2);
  #pragma unroll
  for (int v = 0; v < 8; ++v) {
    int m = v + ((lane >> 4) << 3), n = n0 + (lane & 15);
    Xs[m * 128 + n] = a2[v] + db2[n];
  }
  __syncthreads();
  ln_act_256(Xs, As, dg2, dbe2, da2[0], red);

  // layer 3: 144 output cols = 9 tiles of 16; 8 waves, wave 0 also does tile 8.
  for (int t = wave; t < 9; t += 8) {
    int c0 = t * 16;
    v8f a3 = {};
    #pragma unroll
    for (int kt = 0; kt < 4; ++kt)
      a3 = wmma16(frag_a(As, lane, kt * 32, KPAD),
                  frag_b(dW3h, lane, c0, kt * 32, 128), a3);
    #pragma unroll
    for (int v = 0; v < 8; ++v) {
      int m = v + ((lane >> 4) << 3), n = c0 + (lane & 15);
      float x = a3[v] + db3[n];
      out[(size_t)(rb0 + m) * INSZ + n] = 1.f / (1.f + __expf(-x));
    }
  }
}

// ---------------- kernel 3: projections + contrastive loss ----------------
__global__ __launch_bounds__(128) void proj_loss_kernel(
    const float* __restrict__ ctxsum, const float* __restrict__ currf,
    const float* __restrict__ negf,
    const float* __restrict__ cpW1, const float* __restrict__ cpW2,
    const float* __restrict__ cpb1, const float* __restrict__ cpb2,
    const float* __restrict__ uW1, const float* __restrict__ uW2,
    const float* __restrict__ ub1, const float* __restrict__ ub2,
    float* __restrict__ loss) {
  __shared__ float xv[4][128];
  __shared__ float h1[4][64];
  __shared__ float o2[4][32];
  __shared__ float dots[3];
  const int tid = threadIdx.x;
  float lsum = 0.f;
  for (int bi = 0; bi < 16; ++bi) {
    int b = blockIdx.x * 16 + bi;
    xv[0][tid] = ctxsum[(size_t)b * 128 + tid] * (1.f / 64.f);  // context vec
    xv[1][tid] = currf[(size_t)b * 128 + tid];                  // current feat
    xv[2][tid] = negf[(size_t)(b * 2 + 0) * 128 + tid];
    xv[3][tid] = negf[(size_t)(b * 2 + 1) * 128 + tid];
    __syncthreads();
    for (int t = tid; t < 256; t += 128) {      // layer1: 4 vecs x 64 outs
      int v = t >> 6, j = t & 63;
      const float* W  = (v == 0) ? cpW1 : uW1;
      const float* bb = (v == 0) ? cpb1 : ub1;
      float s = bb[j];
      for (int k = 0; k < 128; ++k) s += xv[v][k] * W[j * 128 + k];
      h1[v][j] = s > 0.f ? s : 0.f;
    }
    __syncthreads();
    {                                           // layer2: 4 vecs x 32 outs
      int v = tid >> 5, j = tid & 31;
      const float* W  = (v == 0) ? cpW2 : uW2;
      const float* bb = (v == 0) ? cpb2 : ub2;
      float s = bb[j];
      for (int k = 0; k < 64; ++k) s += h1[v][k] * W[j * 64 + k];
      o2[v][j] = s;
    }
    __syncthreads();
    if (tid < 3) {
      float s = 0.f;
      for (int k = 0; k < 32; ++k) s += o2[0][k] * o2[tid + 1][k];
      dots[tid] = s;
    }
    __syncthreads();
    if (tid == 0) {
      float pos = __expf(dots[0]) * 0.5f;
      float e0  = __expf(dots[1]) * 0.5f;
      float e1  = __expf(dots[2]) * 0.5f;
      lsum += -__logf(pos / (pos + e0 + e1));
    }
    __syncthreads();
  }
  if (tid == 0) atomicAdd(loss, lsum);
}

__global__ void finish_kernel(const float* loss, float* out) {
  out[(size_t)BSZ * INSZ] = loss[0] * (1.f / (float)BSZ);
}

// ---------------- host launch ----------------
extern "C" void kernel_launch(void* const* d_in, const int* in_sizes, int n_in,
                              void* d_out, int out_size, void* d_ws, size_t ws_size,
                              hipStream_t stream) {
  (void)in_sizes; (void)n_in; (void)out_size; (void)ws_size;
  char* ws = (char*)d_ws;
  float*     ctxsum = (float*)(ws + WS_CTXSUM);
  float*     currf  = (float*)(ws + WS_CURRF);
  float*     negf   = (float*)(ws + WS_NEGF);
  float*     lossw  = (float*)(ws + WS_LOSS);
  _Float16*  eW1h   = (_Float16*)(ws + WS_EW1H);
  _Float16*  eW2h   = (_Float16*)(ws + WS_EW2H);
  _Float16*  eW3h   = (_Float16*)(ws + WS_EW3H);
  _Float16*  wvh    = (_Float16*)(ws + WS_WVH);
  _Float16*  dW1h   = (_Float16*)(ws + WS_DW1H);
  _Float16*  dW2h   = (_Float16*)(ws + WS_DW2H);
  _Float16*  dW3h   = (_Float16*)(ws + WS_DW3H);
  float* out = (float*)d_out;
  const float* F = nullptr; (void)F;
  #define INF(i) ((const float*)d_in[(i)])

  // 0) zero accumulators + convert weights to f16
  {
    long total = (long)BSZ * 128 + 128 * KPAD + 5 * 16384 + INSZ * 128;
    int blocks = (int)((total + 255) / 256);
    prep_kernel<<<blocks, 256, 0, stream>>>(
        INF(IN_ENC_W1), INF(IN_ENC_W2), INF(IN_ENC_W3), INF(IN_ATTN_WV),
        INF(IN_DEC_W1), INF(IN_DEC_W2), INF(IN_DEC_W3),
        eW1h, eW2h, eW3h, wvh, dW1h, dW2h, dW3h, ctxsum, lossw);
  }
  // 1) fused encoder (+ attn-k transform and mean accumulation for ctx)
  enc_kernel<<<ENC_BLOCKS, 256, 0, stream>>>(
      INF(IN_CTX), INF(IN_CUR), INF(IN_NEG),
      eW1h, eW2h, eW3h, wvh,
      INF(IN_ENC_B1), INF(IN_ENC_G1), INF(IN_ENC_BE1), INF(IN_ENC_A1),
      INF(IN_ENC_B2), INF(IN_ENC_G2), INF(IN_ENC_BE2), INF(IN_ENC_A2),
      INF(IN_ENC_B3), INF(IN_ATTN_BV),
      ctxsum, currf, negf);
  // 2) decoder -> sigmoid(pred)
  dec_kernel<<<BSZ / 16, 256, 0, stream>>>(
      ctxsum, currf, dW1h, dW2h, dW3h,
      INF(IN_NORM_G), INF(IN_NORM_B),
      INF(IN_DEC_B1), INF(IN_DEC_G1), INF(IN_DEC_BE1), INF(IN_DEC_A1),
      INF(IN_DEC_B2), INF(IN_DEC_G2), INF(IN_DEC_BE2), INF(IN_DEC_A2),
      INF(IN_DEC_B3), out);
  // 3) projections + contrastive loss partials
  proj_loss_kernel<<<BSZ / 16, 128, 0, stream>>>(
      ctxsum, currf, negf,
      INF(IN_CTXP_W1), INF(IN_CTXP_W2), INF(IN_CTXP_B1), INF(IN_CTXP_B2),
      INF(IN_CURP_W1), INF(IN_CURP_W2), INF(IN_CURP_B1), INF(IN_CURP_B2),
      lossw);
  // 4) finalize loss
  finish_kernel<<<1, 1, 0, stream>>>(lossw, out);
  #undef INF
}
